// PatchAbstraction_7730941133029
// MI455X (gfx1250) — matrix operand order repaired and checked by reading, not compiled, not run on hardware
//
#include <hip/hip_runtime.h>
#include <hip/hip_bf16.h>

// ---------------------------------------------------------------------------
// PatchAbstraction for MI455X (gfx1250, wave32, WMMA)
//
// Input index map (setup_inputs() dict order, tuples flattened):
//  0 xyz[B,N,3]  1 feature[B,N,64]
//  2..4 conv_ws  5..7 conv_bs  8..10 bn_g  11..13 bn_b  14..16 bn_m  17..19 bn_v
//  20 res_w  21 res_b  22..25 res_bn(g,b,m,v)
//  26 fc1_w  27 fc1_b  28 fcv_w  29 fcv_b  30 fc2_w  31 fc2_b
//  32 sim_alpha  33 sim_beta
//
// d_out = concat(centroid_xyz [8,1024,3], max_p [8,1024,256], avg_p [8,1024,256])
// ---------------------------------------------------------------------------

#define BB 8
#define NN 8192
#define SS 1024
#define KK 32
#define CF 64
#define DD 256
#define HH 4
#define HD 24
#define HF 96           // HEADS*HEAD_DIM
#define RR (BB*SS*KK)   // 262144 rows of grouped features

typedef __attribute__((ext_vector_type(16))) __bf16 v16bf;
typedef __attribute__((ext_vector_type(8)))  float  v8f;

union FragU { v16bf v; unsigned u[8]; };

__device__ __forceinline__ unsigned short f2bf(float f) {
    unsigned u = __builtin_bit_cast(unsigned, f);
    unsigned r = (u + 0x7fffu + ((u >> 16) & 1u)) >> 16;
    return (unsigned short)r;
}
__device__ __forceinline__ float bf2f(unsigned short h) {
    unsigned u = ((unsigned)h) << 16;
    return __builtin_bit_cast(float, u);
}

// ---------------------------------------------------------------------------
// 1) Furthest-point sampling: one 1024-thread block per batch. Each thread
//    owns 8 points (strided), dmin lives in registers. Deterministic argmax
//    with lowest-index tie-break (matches jnp.argmax).
// ---------------------------------------------------------------------------
__global__ __launch_bounds__(1024)
void fps_kernel(const float* __restrict__ xyz, int* __restrict__ fpsIdx,
                float* __restrict__ centOut) {
    const int b = blockIdx.x, tid = threadIdx.x;
    const float* base = xyz + (size_t)b * NN * 3;
    float px[8], py[8], pz[8], dmin[8];
#pragma unroll
    for (int j = 0; j < 8; ++j) {
        int n = tid + j * 1024;
        px[j] = base[n * 3 + 0]; py[j] = base[n * 3 + 1]; pz[j] = base[n * 3 + 2];
        dmin[j] = 1e10f;
    }
    __shared__ float wv[32]; __shared__ int wi[32];
    __shared__ float lx, ly, lz;
    if (tid == 0) {
        fpsIdx[b * SS] = 0;
        lx = base[0]; ly = base[1]; lz = base[2];
        size_t o = (size_t)b * SS * 3;
        centOut[o] = lx; centOut[o + 1] = ly; centOut[o + 2] = lz;
    }
    __syncthreads();
    for (int i = 1; i < SS; ++i) {
        const float cx = lx, cy = ly, cz = lz;
        float bv = -1.0f; int bi = 0;
#pragma unroll
        for (int j = 0; j < 8; ++j) {
            float dx = px[j] - cx, dy = py[j] - cy, dz = pz[j] - cz;
            float d = dx * dx + dy * dy + dz * dz;
            float dm = fminf(dmin[j], d);
            dmin[j] = dm;
            int n = tid + j * 1024;
            if (dm > bv || (dm == bv && n < bi)) { bv = dm; bi = n; }
        }
        // wave32 reduce (max value, min index on tie)
        for (int off = 16; off; off >>= 1) {
            float ov = __shfl_down(bv, off);
            int   oi = __shfl_down(bi, off);
            if (ov > bv || (ov == bv && oi < bi)) { bv = ov; bi = oi; }
        }
        if ((tid & 31) == 0) { wv[tid >> 5] = bv; wi[tid >> 5] = bi; }
        __syncthreads();
        if (tid < 32) {
            float v2 = wv[tid]; int i2 = wi[tid];
            for (int off = 16; off; off >>= 1) {
                float ov = __shfl_down(v2, off);
                int   oi = __shfl_down(i2, off);
                if (ov > v2 || (ov == v2 && oi < i2)) { v2 = ov; i2 = oi; }
            }
            if (tid == 0) {
                fpsIdx[b * SS + i] = i2;
                float nx = base[i2 * 3], ny = base[i2 * 3 + 1], nz = base[i2 * 3 + 2];
                lx = nx; ly = ny; lz = nz;
                size_t o = ((size_t)b * SS + i) * 3;
                centOut[o] = nx; centOut[o + 1] = ny; centOut[o + 2] = nz;
            }
        }
        __syncthreads();
    }
}

// ---------------------------------------------------------------------------
// 2) kNN: one block per centroid; all 8192 distances in LDS, then 32
//    deterministic argmin+mark passes (matches top_k(-d) ordering).
// ---------------------------------------------------------------------------
__global__ __launch_bounds__(256)
void knn_kernel(const float* __restrict__ xyz, const int* __restrict__ fpsIdx,
                int* __restrict__ knnIdx) {
    const int bs = blockIdx.x;          // b*S + s
    const int b = bs >> 10;
    __shared__ float dist[NN];
    __shared__ float wv[8]; __shared__ int wi[8];
    const int tid = threadIdx.x;
    const float* base = xyz + (size_t)b * NN * 3;
    const int cidx = fpsIdx[bs];
    const float cx = base[cidx * 3], cy = base[cidx * 3 + 1], cz = base[cidx * 3 + 2];
    for (int n = tid; n < NN; n += 256) {
        float dx = base[n * 3] - cx, dy = base[n * 3 + 1] - cy, dz = base[n * 3 + 2] - cz;
        dist[n] = dx * dx + dy * dy + dz * dz;
    }
    __syncthreads();
    for (int kk = 0; kk < KK; ++kk) {
        float bv = 3.0e38f; int bi = 0x7fffffff;
        for (int n = tid; n < NN; n += 256) {
            float v = dist[n];
            if (v < bv || (v == bv && n < bi)) { bv = v; bi = n; }
        }
        for (int off = 16; off; off >>= 1) {
            float ov = __shfl_down(bv, off);
            int   oi = __shfl_down(bi, off);
            if (ov < bv || (ov == bv && oi < bi)) { bv = ov; bi = oi; }
        }
        if ((tid & 31) == 0) { wv[tid >> 5] = bv; wi[tid >> 5] = bi; }
        __syncthreads();
        if (tid == 0) {
            float fb = wv[0]; int fi = wi[0];
            for (int j = 1; j < 8; ++j)
                if (wv[j] < fb || (wv[j] == fb && wi[j] < fi)) { fb = wv[j]; fi = wi[j]; }
            knnIdx[(size_t)bs * KK + kk] = fi;
            dist[fi] = 3.0e38f;
        }
        __syncthreads();
    }
}

// ---------------------------------------------------------------------------
// 3) Gather + group -> Xg bf16 [R,128]: cols 0..63 centroid feat, 64..127 diff
// ---------------------------------------------------------------------------
__global__ __launch_bounds__(256)
void gather_kernel(const float* __restrict__ feat, const int* __restrict__ fpsIdx,
                   const int* __restrict__ knnIdx, unsigned short* __restrict__ Xg) {
    long long e = (long long)blockIdx.x * 256 + threadIdx.x;   // < R*128
    int c = (int)(e & 127);
    long long r = e >> 7;                                       // row in [0,R)
    long long bs = r >> 5;                                      // b*S+s
    int b = (int)(bs >> 10);
    int cidx = fpsIdx[bs];
    int cc = c & 63;
    float cf = feat[((size_t)b * NN + cidx) * CF + cc];
    float val;
    if (c < 64) {
        val = cf;
    } else {
        int nidx = knnIdx[r];
        val = feat[((size_t)b * NN + nidx) * CF + cc] - cf;
    }
    Xg[(size_t)r * 128 + c] = f2bf(val);
}

// ---------------------------------------------------------------------------
// 4) Generic bf16 WMMA GEMM:  out = epi(A[M,K](bf16) x W[N,K]^T(f32))
//    Block tile 256x64x32; 8 waves; each wave owns a 32x64 register tile
//    (2 A-frags x 4 B-frags -> 8 x v_wmma_f32_16x16x32_bf16 per k-step,
//    issued back-to-back after all fragment loads so the XDL pipe stays fed).
//    Epilogue: val = acc*scale + shift (+resid) (relu); scale/shift fold the
//    conv bias + eval-mode BN. Optional f32/bf16 outputs.
// ---------------------------------------------------------------------------
__global__ __launch_bounds__(256)
void gemm_bf16_wmma(const unsigned short* __restrict__ A, const float* __restrict__ W,
                    float* __restrict__ outF, unsigned short* __restrict__ outB,
                    int M, int Nn, int K,
                    const float* __restrict__ bias,
                    const float* __restrict__ bng, const float* __restrict__ bnb,
                    const float* __restrict__ bnm, const float* __restrict__ bnv,
                    const float* __restrict__ resid, int relu) {
    (void)M;
    const int tid = threadIdx.x;
    const int wave = tid >> 5, lane = tid & 31;
    const int rowBlk = blockIdx.x * 256;
    const int colBlk = blockIdx.y * 64;

    __shared__ unsigned short As[256][48];   // 32 used + pad, 16B-aligned rows
    __shared__ unsigned short Bs[64][48];

    v8f acc[2][4];
#pragma unroll
    for (int mt = 0; mt < 2; ++mt)
#pragma unroll
        for (int nt = 0; nt < 4; ++nt)
#pragma unroll
            for (int j = 0; j < 8; ++j) acc[mt][nt][j] = 0.0f;

    for (int kk = 0; kk < K; kk += 32) {
        // --- stage A tile (256x32 bf16): one 64B row per thread ---
        {
            const unsigned short* src = A + (size_t)(rowBlk + tid) * K + kk;
            uint4 d0 = *(const uint4*)(src);
            uint4 d1 = *(const uint4*)(src + 8);
            uint4 d2 = *(const uint4*)(src + 16);
            uint4 d3 = *(const uint4*)(src + 24);
            *(uint4*)&As[tid][0]  = d0;
            *(uint4*)&As[tid][8]  = d1;
            *(uint4*)&As[tid][16] = d2;
            *(uint4*)&As[tid][24] = d3;
            if (kk + 32 < K)
                __builtin_prefetch(src + 32, 0, 3);   // global_prefetch next k-slab
        }
        // --- stage B tile (64x32 bf16 from f32 W[N,K]) : 4 threads/row ---
        {
            int r = tid >> 2;
            int kc = (tid & 3) * 8;
            int ng = colBlk + r;
            unsigned short tmp[8];
            if (ng < Nn) {
                const float* src = W + (size_t)ng * K + kk + kc;
#pragma unroll
                for (int j = 0; j < 8; ++j) tmp[j] = f2bf(src[j]);
            } else {
#pragma unroll
                for (int j = 0; j < 8; ++j) tmp[j] = 0;
            }
            *(uint4*)&Bs[r][kc] = *(const uint4*)tmp;
        }
        __syncthreads();

        // --- A fragments (16x32, ISA 16-bit A layout: lanes 0-15 K 0..7/16..23,
        //     lanes 16-31 K 8..15/24..31; m = lane&15) ---
        FragU afrag[2];
        {
            const int m = lane & 15;
            const int kb = (lane < 16) ? 0 : 8;
#pragma unroll
            for (int mt = 0; mt < 2; ++mt) {
                const unsigned short* ar = &As[wave * 32 + mt * 16 + m][0];
#pragma unroll
                for (int j = 0; j < 4; ++j) {
                    afrag[mt].u[j]     = *(const unsigned*)(ar + kb + 2 * j);
                    afrag[mt].u[4 + j] = *(const unsigned*)(ar + kb + 16 + 2 * j);
                }
            }
        }
        // --- B fragments (32x16, lanes 0-15 K 0..7/16..23, lanes 16-31 K 8..15/24..31) ---
        FragU bfrag[4];
        {
            const int nl = lane & 15;
            const int kbb = (lane < 16) ? 0 : 8;
#pragma unroll
            for (int nt = 0; nt < 4; ++nt) {
                const unsigned short* br = &Bs[nt * 16 + nl][0];
#pragma unroll
                for (int j = 0; j < 8; ++j)
                    bfrag[nt].u[j] = *(const unsigned*)(br + kbb + 2 * j + (j >= 4 ? 8 : 0));
            }
        }
        // --- 8 back-to-back WMMAs on the register tile ---
#pragma unroll
        for (int mt = 0; mt < 2; ++mt)
#pragma unroll
            for (int nt = 0; nt < 4; ++nt)
                acc[mt][nt] = __builtin_amdgcn_wmma_f32_16x16x32_bf16(
                    false, afrag[mt].v, false, bfrag[nt].v, (short)0,
                    acc[mt][nt], false, false);
        __syncthreads();
    }

    // --- epilogue: C/D 16x16 f32 layout (n = lane&15, m = r + 8*(lane>=16)) ---
    const int mhi = (lane >> 4) * 8;
#pragma unroll
    for (int nt = 0; nt < 4; ++nt) {
        int ng = colBlk + nt * 16 + (lane & 15);
        if (ng >= Nn) continue;
        float scale = 1.0f;
        float shift = bias ? bias[ng] : 0.0f;
        if (bng) {
            scale = bng[ng] * rsqrtf(bnv[ng] + 1e-5f);
            shift = (shift - bnm[ng]) * scale + bnb[ng];
        }
#pragma unroll
        for (int mt = 0; mt < 2; ++mt) {
#pragma unroll
            for (int r = 0; r < 8; ++r) {
                size_t row = (size_t)rowBlk + wave * 32 + mt * 16 + r + mhi;
                float v = acc[mt][nt][r] * scale + shift;
                if (resid) v += resid[row * Nn + ng];
                if (relu) v = fmaxf(v, 0.0f);
                if (outF) outF[row * Nn + ng] = v;
                if (outB) outB[row * Nn + ng] = f2bf(v);
            }
        }
    }
}

// ---------------------------------------------------------------------------
// 5) Max / avg pool over K=32 neighbors
// ---------------------------------------------------------------------------
__global__ __launch_bounds__(256)
void pool_kernel(const unsigned short* __restrict__ H3, float* __restrict__ maxF,
                 unsigned short* __restrict__ maxB, float* __restrict__ avgOut) {
    int gid = blockIdx.x * 256 + threadIdx.x;    // bs*256 + c
    int bs = gid >> 8; int c = gid & 255;
    const unsigned short* p = H3 + ((size_t)bs * KK) * DD + c;
    float mx = -3.0e38f, sm = 0.0f;
#pragma unroll
    for (int k = 0; k < KK; ++k) {
        float v = bf2f(p[(size_t)k * DD]);
        mx = fmaxf(mx, v); sm += v;
    }
    maxF[gid] = mx;
    maxB[gid] = f2bf(mx);
    avgOut[gid] = sm * (1.0f / 32.0f);
}

// ---------------------------------------------------------------------------
// 6) ContextCluster per-(b,h) math (deterministic partial-sum reductions)
// ---------------------------------------------------------------------------
__global__ __launch_bounds__(256)
void cluster_kernel(const float* __restrict__ q, const float* __restrict__ v,
                    const float* __restrict__ alpha, const float* __restrict__ beta,
                    float* __restrict__ coF, unsigned short* __restrict__ coB) {
    const int bh = blockIdx.x;
    const int b = bh >> 2, h = bh & 3;
    const int tid = threadIdx.x;
    __shared__ float cen[HD], vcen[HD], agg[HD], chat[HD], obase[HD];
    __shared__ float sims[SS];
    __shared__ float partA[256], partB[256];
    __shared__ float simsumS;
    const float* qb = q + (size_t)b * SS * HF + h * HD;
    const float* vb = v + (size_t)b * SS * HF + h * HD;
    const int d = tid >> 3, p = tid & 7;

    // center / v_center means
    if (d < HD) {
        float sq = 0.0f, sv = 0.0f;
        for (int s = p; s < SS; s += 8) {
            sq += qb[(size_t)s * HF + d];
            sv += vb[(size_t)s * HF + d];
        }
        partA[tid] = sq; partB[tid] = sv;
    } else { partA[tid] = 0.0f; partB[tid] = 0.0f; }
    __syncthreads();
    if (d < HD && p == 0) {
        float sq = 0.0f, sv = 0.0f;
        for (int j = 0; j < 8; ++j) { sq += partA[(d << 3) + j]; sv += partB[(d << 3) + j]; }
        cen[d] = sq * (1.0f / SS); vcen[d] = sv * (1.0f / SS);
    }
    __syncthreads();
    if (tid == 0) {
        float nn2 = 0.0f;
        for (int i = 0; i < HD; ++i) nn2 += cen[i] * cen[i];
        float inv = 1.0f / fmaxf(sqrtf(nn2), 1e-12f);
        for (int i = 0; i < HD; ++i) chat[i] = cen[i] * inv;
    }
    __syncthreads();
    // similarities
    const float sa = alpha[0], sb = beta[0];
    for (int s = tid; s < SS; s += 256) {
        float dot = 0.0f, nq = 0.0f;
        for (int i = 0; i < HD; ++i) {
            float qv = qb[(size_t)s * HF + i];
            dot += qv * chat[i]; nq += qv * qv;
        }
        float cosv = dot / fmaxf(sqrtf(nq), 1e-12f);
        sims[s] = 1.0f / (1.0f + __expf(-(sb + sa * cosv)));
    }
    __syncthreads();
    // agg = sum_s sim*v ; simsum = sum_s sim
    if (d < HD) {
        float s_ = 0.0f;
        for (int s = p; s < SS; s += 8) s_ += sims[s] * vb[(size_t)s * HF + d];
        partA[tid] = s_;
    } else partA[tid] = 0.0f;
    { float ss = 0.0f; for (int s = tid; s < SS; s += 256) ss += sims[s]; partB[tid] = ss; }
    __syncthreads();
    if (d < HD && p == 0) {
        float s_ = 0.0f;
        for (int j = 0; j < 8; ++j) s_ += partA[(d << 3) + j];
        agg[d] = s_;
    }
    if (tid == 0) {
        float ss = 0.0f;
        for (int j = 0; j < 256; ++j) ss += partB[j];
        simsumS = ss;
    }
    __syncthreads();
    if (tid == 0) {
        float inv = 1.0f / (simsumS + 1.0f);
        for (int i = 0; i < HD; ++i) obase[i] = (agg[i] + vcen[i]) * inv;
    }
    __syncthreads();
    for (int s = tid; s < SS; s += 256) {
        float sim = sims[s];
        size_t base = ((size_t)(b * SS + s)) * HF + h * HD;
        for (int i = 0; i < HD; ++i) {
            float val = obase[i] * sim;
            coF[base + i] = val;
            coB[base + i] = f2bf(val);
        }
    }
}

// ---------------------------------------------------------------------------
extern "C" void kernel_launch(void* const* d_in, const int* in_sizes, int n_in,
                              void* d_out, int out_size, void* d_ws, size_t ws_size,
                              hipStream_t stream) {
    (void)in_sizes; (void)n_in; (void)out_size; (void)ws_size;

    const float* xyz    = (const float*)d_in[0];
    const float* feat   = (const float*)d_in[1];
    const float* convW[3] = { (const float*)d_in[2], (const float*)d_in[3], (const float*)d_in[4] };
    const float* convB[3] = { (const float*)d_in[5], (const float*)d_in[6], (const float*)d_in[7] };
    const float* bnG[3]   = { (const float*)d_in[8], (const float*)d_in[9], (const float*)d_in[10] };
    const float* bnB[3]   = { (const float*)d_in[11], (const float*)d_in[12], (const float*)d_in[13] };
    const float* bnM[3]   = { (const float*)d_in[14], (const float*)d_in[15], (const float*)d_in[16] };
    const float* bnV[3]   = { (const float*)d_in[17], (const float*)d_in[18], (const float*)d_in[19] };
    const float* resW  = (const float*)d_in[20];
    const float* resB  = (const float*)d_in[21];
    const float* resG  = (const float*)d_in[22];
    const float* resBb = (const float*)d_in[23];
    const float* resM  = (const float*)d_in[24];
    const float* resV  = (const float*)d_in[25];
    const float* fc1W  = (const float*)d_in[26];
    const float* fc1B  = (const float*)d_in[27];
    const float* fcvW  = (const float*)d_in[28];
    const float* fcvB  = (const float*)d_in[29];
    const float* fc2W  = (const float*)d_in[30];
    const float* fc2B  = (const float*)d_in[31];
    const float* simA  = (const float*)d_in[32];
    const float* simBt = (const float*)d_in[33];

    float* outCent = (float*)d_out;                               // [B,S,3]
    float* outMax  = outCent + (size_t)BB * SS * 3;               // [B,S,256]
    float* outAvg  = outMax + (size_t)BB * SS * DD;               // [B,S,256]

    // --- workspace carve-up (256B aligned) ---
    char* ws = (char*)d_ws;
    size_t off = 0;
    auto carve = [&](size_t bytes) { char* p = ws + off; off += (bytes + 255) & ~(size_t)255; return p; };
    int*            fpsIdx = (int*)carve((size_t)BB * SS * 4);
    int*            knnIdx = (int*)carve((size_t)BB * SS * KK * 4);
    unsigned short* Xg     = (unsigned short*)carve((size_t)RR * 128 * 2);
    unsigned short* H1     = (unsigned short*)carve((size_t)RR * 128 * 2);
    unsigned short* H3     = (unsigned short*)carve((size_t)RR * 256 * 2);
    float*          maxF   = (float*)carve((size_t)BB * SS * DD * 4);
    unsigned short* maxB16 = (unsigned short*)carve((size_t)BB * SS * DD * 2);
    float*          xF     = (float*)carve((size_t)BB * SS * DD * 4);
    unsigned short* xB16   = (unsigned short*)carve((size_t)BB * SS * DD * 2);
    float*          qF     = (float*)carve((size_t)BB * SS * HF * 4);
    float*          vF     = (float*)carve((size_t)BB * SS * HF * 4);
    float*          coF    = (float*)carve((size_t)BB * SS * HF * 4);
    unsigned short* coB    = (unsigned short*)carve((size_t)BB * SS * HF * 2);

    // 1) FPS + centroid_xyz output
    fps_kernel<<<BB, 1024, 0, stream>>>(xyz, fpsIdx, outCent);
    // 2) kNN
    knn_kernel<<<BB * SS, 256, 0, stream>>>(xyz, fpsIdx, knnIdx);
    // 3) gather + group (bf16)
    gather_kernel<<<(int)((size_t)RR * 128 / 256), 256, 0, stream>>>(feat, fpsIdx, knnIdx, Xg);
    // 4) conv stack (WMMA bf16): 128->128->128->256 with bias+BN+ReLU fused
    gemm_bf16_wmma<<<dim3(RR / 256, 2), 256, 0, stream>>>(
        Xg, convW[0], nullptr, H1, RR, 128, 128,
        convB[0], bnG[0], bnB[0], bnM[0], bnV[0], nullptr, 1);
    gemm_bf16_wmma<<<dim3(RR / 256, 2), 256, 0, stream>>>(
        H1, convW[1], nullptr, Xg, RR, 128, 128,
        convB[1], bnG[1], bnB[1], bnM[1], bnV[1], nullptr, 1);
    gemm_bf16_wmma<<<dim3(RR / 256, 4), 256, 0, stream>>>(
        Xg, convW[2], nullptr, H3, RR, 256, 128,
        convB[2], bnG[2], bnB[2], bnM[2], bnV[2], nullptr, 1);
    // 5) max/avg pool over K (avg_p straight to d_out)
    pool_kernel<<<(BB * SS * DD) / 256, 256, 0, stream>>>(H3, maxF, maxB16, outAvg);
    // 6) ResMLP: relu(bn(maxp @ resW^T + resB) + maxp)
    gemm_bf16_wmma<<<dim3(BB * SS / 256, 4), 256, 0, stream>>>(
        maxB16, resW, xF, xB16, BB * SS, 256, 256,
        resB, resG, resBb, resM, resV, maxF, 1);
    // 7) q / v projections
    gemm_bf16_wmma<<<dim3(BB * SS / 256, 2), 256, 0, stream>>>(
        xB16, fc1W, qF, nullptr, BB * SS, HF, 256,
        fc1B, nullptr, nullptr, nullptr, nullptr, nullptr, 0);
    gemm_bf16_wmma<<<dim3(BB * SS / 256, 2), 256, 0, stream>>>(
        xB16, fcvW, vF, nullptr, BB * SS, HF, 256,
        fcvB, nullptr, nullptr, nullptr, nullptr, nullptr, 0);
    // 8) context cluster math
    cluster_kernel<<<BB * HH, 256, 0, stream>>>(qF, vF, simA, simBt, coF, coB);
    // 9) fc2 + residual x -> final max_p (to d_out)
    gemm_bf16_wmma<<<dim3(BB * SS / 256, 4), 256, 0, stream>>>(
        coB, fc2W, outMax, nullptr, BB * SS, 256, HF,
        fc2B, nullptr, nullptr, nullptr, nullptr, xF, 0);
}
